// EfficientSVDNet_74534862455116
// MI455X (gfx1250) — compile-verified, matrix operand order checked by reading, not run on hardware
//
#include <hip/hip_runtime.h>
#include <hip/hip_bf16.h>

typedef _Float16 h16;
typedef __attribute__((ext_vector_type(8)))  _Float16 v8h;
typedef __attribute__((ext_vector_type(16))) _Float16 v16h;
typedef __attribute__((ext_vector_type(8)))  float    v8f;

#define WMMA_F16(A,B,C) __builtin_amdgcn_wmma_f32_16x16x32_f16(false,(A),false,(B),(short)0,(C),false,false)

// Fragment loader for the CDNA5 16-bit A/B WMMA layout.
// Lane l (half = l>>4) element e maps to k = (e&7) + half*8 + (e>=8)*16.
// With K contiguous in memory this is two 16B loads: [half*8 .. +7] and [16+half*8 .. +7].
__device__ __forceinline__ v16h frag_rowK(const h16* p, int half) {
  v8h lo = *(const v8h*)(p + half * 8);
  v8h hi = *(const v8h*)(p + 16 + half * 8);
  v16h r;
#pragma unroll
  for (int i = 0; i < 8; ++i) { r[i] = lo[i]; r[8 + i] = hi[i]; }
  return r;
}

// CDNA5 async copy: global -> LDS, 16 bytes per active lane, tracked by ASYNCcnt.
// LDS operand is the wave-relative LDS byte address = low 32 bits of the generic pointer.
__device__ __forceinline__ void async_ld_b128(void* lds_generic, const void* gptr) {
  unsigned loff = (unsigned)(unsigned long long)lds_generic;
  asm volatile("global_load_async_to_lds_b128 %0, %1, off"
               :: "v"(loff), "v"(gptr)
               : "memory");
}
__device__ __forceinline__ void wait_async0() {
  asm volatile("s_wait_asynccnt 0" ::: "memory");
}

// ---------------------------------------------------------------- prep kernels

__global__ void k_bnprep(const float* g, const float* be, const float* mn, const float* vr,
                         float* sc, float* bi, int n) {
  int i = blockIdx.x * 64 + threadIdx.x;
  if (i < n) {
    float s = g[i] * rsqrtf(vr[i] + 1e-5f);
    sc[i] = s;
    bi[i] = be[i] - mn[i] * s;
  }
}

// w[O][I][3][3] f32 -> dst[O][9*I] f16 with k = tap*I + ci (tap = kh*3+kw)
__global__ void k_pack3x3(const float* w, h16* d, int O, int I) {
  int i = blockIdx.x * 256 + threadIdx.x;
  if (i < O * I * 9) {
    int o = i / (I * 9), r = i % (I * 9), t = r / I, ci = r % I;
    d[i] = (h16)w[(o * I + ci) * 9 + t];
  }
}

__global__ void k_cvt(const float* s, h16* d, int n) {
  int i = blockIdx.x * 256 + threadIdx.x;
  if (i < n) d[i] = (h16)s[i];
}

// -------------------------------------------------- rb1 conv1 + shortcut (VALU, K=18)

__global__ __launch_bounds__(256) void k_rb1a(const float* H, const float* w1, const float* wsc,
                                              const float* bn1sc, const float* bn1bi,
                                              const float* bscsc, const float* bscbi,
                                              h16* act1, h16* sc1) {
  long gid = (long)blockIdx.x * 256 + threadIdx.x;   // B*64*64*32 threads
  int co = gid & 31;
  long pix = gid >> 5;
  int wo = pix & 63;
  long r2 = pix >> 6;
  int ho = r2 & 63;
  int b  = (int)(r2 >> 6);
  float acc = 0.f;
#pragma unroll
  for (int kh = 0; kh < 3; ++kh) {
    int ih = 2 * ho + kh - 1;
    if (ih < 0 || ih >= 128) continue;
#pragma unroll
    for (int kw = 0; kw < 3; ++kw) {
      int iw = 2 * wo + kw - 1;
      if (iw < 0 || iw >= 128) continue;
      const float* px = H + (((long)b * 128 + ih) * 128 + iw) * 2;
      acc += px[0] * w1[((co * 2 + 0) * 3 + kh) * 3 + kw]
           + px[1] * w1[((co * 2 + 1) * 3 + kh) * 3 + kw];
    }
  }
  long o = (((long)(b * 64 + ho) * 64) + wo) * 32 + co;
  act1[o] = (h16)fmaxf(acc * bn1sc[co] + bn1bi[co], 0.f);
  const float* pc = H + (((long)b * 128 + 2 * ho) * 128 + 2 * wo) * 2;
  float sv = pc[0] * wsc[co * 2] + pc[1] * wsc[co * 2 + 1];
  sc1[o] = (h16)(sv * bscsc[co] + bscbi[co]);
}

// -------------------------------------------------- rb1 conv2 (WMMA K=288 N=32) + bn + sc + relu

__global__ __launch_bounds__(256) void k_rb1b(const h16* act1, const h16* sc1, const h16* w2t,
                                              const float* bnsc, const float* bnbi, h16* act2) {
  int bh = blockIdx.x, b = bh >> 6, h = bh & 63;
  __shared__ __attribute__((aligned(16))) h16 tile[3][66][32];
  int tid = threadIdx.x;
  __builtin_prefetch(w2t + tid * 36, 0, 1);          // global_prefetch_b8 on weight panel
  // async-stage the 3x66x32 f16 halo tile: 16B chunks, zero-fill halo
  for (int idx = tid; idx < 3 * 66 * 4; idx += 256) {
    int ch = idx & 3, rest = idx >> 2;
    int w = rest % 66, r = rest / 66;
    int ih = h + r - 1, iw = w - 1;
    h16* dst = &tile[r][w][ch * 8];
    if (ih >= 0 && ih < 64 && iw >= 0 && iw < 64) {
      async_ld_b128(dst, act1 + (((long)(b * 64 + ih) * 64) + iw) * 32 + ch * 8);
    } else {
      v8h z = {};
      *(v8h*)dst = z;
    }
  }
  wait_async0();
  __syncthreads();
  int wv = tid >> 5, lane = tid & 31, half = lane >> 4, lm = lane & 15;
  int pb = (wv >> 1) * 16;   // pixel tile (4 tiles of 16 over the 64-wide row)
  int nb = (wv & 1) * 16;    // out-channel tile (2 of 16)
  v8f acc = {};
#pragma unroll
  for (int t = 0; t < 9; ++t) {
    int th = t / 3, tw = t % 3;
    v16h a  = frag_rowK(&tile[th][pb + lm + tw][0], half);
    v16h bf = frag_rowK(w2t + (nb + lm) * 288 + t * 32, half);
    acc = WMMA_F16(a, bf, acc);
  }
  int n = nb + lm;
#pragma unroll
  for (int r = 0; r < 8; ++r) {
    int m = r + 8 * half, w = pb + m;
    long o = (((long)(b * 64 + h) * 64) + w) * 32 + n;
    float v = acc[r] * bnsc[n] + bnbi[n] + (float)sc1[o];
    act2[o] = (h16)fmaxf(v, 0.f);
  }
}

// -------------------------------------------- rb2 conv1 (WMMA K=288 N=64) + fused 1x1 shortcut

__global__ __launch_bounds__(256) void k_rb2a(const h16* act2, const h16* w1t, const h16* wsct,
                                              const float* bn1sc, const float* bn1bi,
                                              const float* bscsc, const float* bscbi,
                                              h16* act3, h16* sc2) {
  int bh = blockIdx.x, b = bh >> 6, h = bh & 63;
  __shared__ __attribute__((aligned(16))) h16 tile[3][66][32];
  int tid = threadIdx.x;
  __builtin_prefetch(w1t + tid * 72, 0, 1);
  for (int idx = tid; idx < 3 * 66 * 4; idx += 256) {
    int ch = idx & 3, rest = idx >> 2;
    int w = rest % 66, r = rest / 66;
    int ih = h + r - 1, iw = w - 1;
    h16* dst = &tile[r][w][ch * 8];
    if (ih >= 0 && ih < 64 && iw >= 0 && iw < 64) {
      async_ld_b128(dst, act2 + (((long)(b * 64 + ih) * 64) + iw) * 32 + ch * 8);
    } else {
      v8h z = {};
      *(v8h*)dst = z;
    }
  }
  wait_async0();
  __syncthreads();
  int wv = tid >> 5, lane = tid & 31, half = lane >> 4, lm = lane & 15;
  int pb = (wv >> 1) * 16;
  int npair = (wv & 1) * 2;          // this wave owns n-tiles {npair, npair+1} of 4
  v8f acc0 = {}, acc1 = {}, as0 = {}, as1 = {};
#pragma unroll
  for (int t = 0; t < 9; ++t) {
    int th = t / 3, tw = t % 3;
    v16h a  = frag_rowK(&tile[th][pb + lm + tw][0], half);
    v16h b0 = frag_rowK(w1t + ((npair + 0) * 16 + lm) * 288 + t * 32, half);
    v16h b1 = frag_rowK(w1t + ((npair + 1) * 16 + lm) * 288 + t * 32, half);
    acc0 = WMMA_F16(a, b0, acc0);
    acc1 = WMMA_F16(a, b1, acc1);
    if (t == 4) {                    // center tap == the pixel itself -> 1x1 shortcut GEMM (K=32)
      v16h c0 = frag_rowK(wsct + ((npair + 0) * 16 + lm) * 32, half);
      v16h c1 = frag_rowK(wsct + ((npair + 1) * 16 + lm) * 32, half);
      as0 = WMMA_F16(a, c0, as0);
      as1 = WMMA_F16(a, c1, as1);
    }
  }
#pragma unroll
  for (int j = 0; j < 2; ++j) {
    int n = (npair + j) * 16 + lm;
    v8f A = j ? acc1 : acc0;
    v8f S = j ? as1 : as0;
#pragma unroll
    for (int r = 0; r < 8; ++r) {
      int m = r + 8 * half, w = pb + m;
      long o = (((long)(b * 64 + h) * 64) + w) * 64 + n;
      act3[o] = (h16)fmaxf(A[r] * bn1sc[n] + bn1bi[n], 0.f);
      sc2[o]  = (h16)(S[r] * bscsc[n] + bscbi[n]);
    }
  }
}

// -------------------------------------------- rb2 conv2 (WMMA K=576 N=64) + bn + sc + relu

__global__ __launch_bounds__(256) void k_rb2b(const h16* act3, const h16* sc2, const h16* w2t,
                                              const float* bnsc, const float* bnbi, h16* feat) {
  int bh = blockIdx.x, b = bh >> 6, h = bh & 63;
  __shared__ __attribute__((aligned(16))) h16 tile[3][66][64];
  int tid = threadIdx.x;
  __builtin_prefetch(w2t + tid * 144, 0, 1);
  for (int idx = tid; idx < 3 * 66 * 8; idx += 256) {
    int ch = idx & 7, rest = idx >> 3;
    int w = rest % 66, r = rest / 66;
    int ih = h + r - 1, iw = w - 1;
    h16* dst = &tile[r][w][ch * 8];
    if (ih >= 0 && ih < 64 && iw >= 0 && iw < 64) {
      async_ld_b128(dst, act3 + (((long)(b * 64 + ih) * 64) + iw) * 64 + ch * 8);
    } else {
      v8h z = {};
      *(v8h*)dst = z;
    }
  }
  wait_async0();
  __syncthreads();
  int wv = tid >> 5, lane = tid & 31, half = lane >> 4, lm = lane & 15;
  int pb = (wv >> 1) * 16, npair = (wv & 1) * 2;
  v8f acc0 = {}, acc1 = {};
#pragma unroll
  for (int t = 0; t < 9; ++t) {
    int th = t / 3, tw = t % 3;
#pragma unroll
    for (int s = 0; s < 2; ++s) {    // K per tap = 64 -> two 32-steps
      v16h a  = frag_rowK(&tile[th][pb + lm + tw][s * 32], half);
      v16h b0 = frag_rowK(w2t + ((npair + 0) * 16 + lm) * 576 + (t * 2 + s) * 32, half);
      v16h b1 = frag_rowK(w2t + ((npair + 1) * 16 + lm) * 576 + (t * 2 + s) * 32, half);
      acc0 = WMMA_F16(a, b0, acc0);
      acc1 = WMMA_F16(a, b1, acc1);
    }
  }
#pragma unroll
  for (int j = 0; j < 2; ++j) {
    int n = (npair + j) * 16 + lm;
    v8f A = j ? acc1 : acc0;
#pragma unroll
    for (int r = 0; r < 8; ++r) {
      int m = r + 8 * half, w = pb + m;
      long o = (((long)(b * 64 + h) * 64) + w) * 64 + n;
      float v = A[r] * bnsc[n] + bnbi[n] + (float)sc2[o];
      feat[o] = (h16)fmaxf(v, 0.f);
    }
  }
}

// ---------------------------------------------------------------- CBAM channel attention

__global__ __launch_bounds__(256) void k_ca(const h16* feat, const float* fc1, const float* fc2,
                                            float* ca) {
  int b = blockIdx.x, tid = threadIdx.x;
  int c = tid & 63, g = tid >> 6;
  const h16* base = feat + (long)b * 4096 * 64;
  float sm = 0.f, mx = -1e30f;
  for (int p = g; p < 4096; p += 4) {
    float v = (float)base[(long)p * 64 + c];
    sm += v; mx = fmaxf(mx, v);
  }
  __shared__ float ssum[4][64], smax[4][64];
  __shared__ float avgv[64], mxv[64], ha[16], hm[16];
  ssum[g][c] = sm; smax[g][c] = mx;
  __syncthreads();
  if (tid < 64) {
    float s = 0.f, m = -1e30f;
    for (int gg = 0; gg < 4; ++gg) { s += ssum[gg][tid]; m = fmaxf(m, smax[gg][tid]); }
    avgv[tid] = s / 4096.f; mxv[tid] = m;
  }
  __syncthreads();
  if (tid < 16) {
    float a = 0.f, m2 = 0.f;
    for (int cc = 0; cc < 64; ++cc) {
      a  += avgv[cc] * fc1[tid * 64 + cc];
      m2 += mxv[cc]  * fc1[tid * 64 + cc];
    }
    ha[tid] = fmaxf(a, 0.f); hm[tid] = fmaxf(m2, 0.f);
  }
  __syncthreads();
  if (tid < 64) {
    float s = 0.f;
    for (int i = 0; i < 16; ++i) s += (ha[i] + hm[i]) * fc2[tid * 16 + i];
    ca[b * 64 + tid] = 1.f / (1.f + __expf(-s));
  }
}

// spatial stats on x = feat*ca
__global__ __launch_bounds__(256) void k_smap(const h16* feat, const float* ca, float* smap) {
  int p = blockIdx.x * 256 + threadIdx.x;
  int b = p >> 12;
  float sm = 0.f, mx = -1e30f;
  for (int c = 0; c < 64; ++c) {
    float v = (float)feat[(long)p * 64 + c] * ca[b * 64 + c];
    sm += v; mx = fmaxf(mx, v);
  }
  smap[p * 2 + 0] = sm / 64.f;
  smap[p * 2 + 1] = mx;
}

// sa = sigmoid(conv3x3(smap)); featA = feat*ca*sa
__global__ __launch_bounds__(256) void k_saap(const h16* feat, const float* ca, const float* smap,
                                              const float* wsa, h16* featA) {
  int p = blockIdx.x * 256 + threadIdx.x;
  int b = p >> 12, hw = p & 4095, h = hw >> 6, w = hw & 63;
  float s = 0.f;
  for (int cc = 0; cc < 2; ++cc)
    for (int dh = 0; dh < 3; ++dh)
      for (int dw = 0; dw < 3; ++dw) {
        int ih = h + dh - 1, iw = w + dw - 1;
        if (ih < 0 || ih >= 64 || iw < 0 || iw >= 64) continue;
        s += smap[(((long)(b << 12) + (ih << 6) + iw) * 2) + cc] * wsa[(cc * 3 + dh) * 3 + dw];
      }
  float sa = 1.f / (1.f + __expf(-s));
  for (int c = 0; c < 64; ++c) {
    float v = (float)feat[(long)p * 64 + c] * ca[b * 64 + c] * sa;
    featA[(long)p * 64 + c] = (h16)v;
  }
}

// ---------------------------------------------------------------- reductions

__global__ void k_rowmean(const h16* featA, h16* rm) {      // grid B*64, 64 thr
  int bh = blockIdx.x, c = threadIdx.x;
  float s = 0.f;
  for (int w = 0; w < 64; ++w) s += (float)featA[((long)bh * 64 + w) * 64 + c];
  rm[bh * 64 + c] = (h16)(s / 64.f);
}
__global__ void k_colmean(const h16* featA, h16* cm) {      // grid B*64 (b,w), 64 thr
  int bw = blockIdx.x, c = threadIdx.x;
  int b = bw >> 6, w = bw & 63;
  float s = 0.f;
  for (int h = 0; h < 64; ++h) s += (float)featA[(((long)(b * 64 + h) * 64) + w) * 64 + c];
  cm[bw * 64 + c] = (h16)(s / 64.f);
}
__global__ void k_gmean(const h16* featA, float* gm) {      // grid B, 64 thr
  int b = blockIdx.x, c = threadIdx.x;
  float s = 0.f;
  for (int p = 0; p < 4096; ++p) s += (float)featA[((long)b * 4096 + p) * 64 + c];
  gm[b * 64 + c] = s / 4096.f;
}

// ---------------------------------------------------------------- s projection

__global__ void k_sproj(const float* gm, const float* ws, const float* sb,
                        const float* bnsc, const float* bnbi, float* S) {
  int b = blockIdx.x, r = threadIdx.x;                      // 16 threads
  float v = sb[r];
  for (int c = 0; c < 64; ++c) v += gm[b * 64 + c] * ws[r * 64 + c];
  v = v * bnsc[r] + bnbi[r];
  float sp = (v > 20.f) ? v : log1pf(__expf(v));
  S[b * 16 + r] = fminf(fmaxf(sp, 1e-6f), 1000.f);
}

// --------------------------------- u/v projection GEMM (4096x64 @ 64x32, WMMA) + bn + complex scatter

__global__ __launch_bounds__(256) void k_uv(const h16* Am, const h16* wt, const float* cb,
                                            const float* bnsc, const float* bnbi, float* outA) {
  int blk = blockIdx.x;                                     // 64 blocks x 64 rows
  int tid = threadIdx.x, wv = tid >> 5, lane = tid & 31, half = lane >> 4, lm = lane & 15;
  int pb = (wv >> 1) * 16, nb = (wv & 1) * 16;
  int rowbase = blk * 64 + pb;
  __builtin_prefetch(wt + tid * 8, 0, 1);
  v8f acc = {};
#pragma unroll
  for (int s = 0; s < 2; ++s) {
    v16h a  = frag_rowK(Am + (long)(rowbase + lm) * 64 + s * 32, half);
    v16h bf = frag_rowK(wt + (nb + lm) * 64 + s * 32, half);
    acc = WMMA_F16(a, bf, acc);
  }
  int n = nb + lm;
#pragma unroll
  for (int r = 0; r < 8; ++r) {
    int row = rowbase + r + 8 * half;
    int b = row >> 6, x = row & 63;
    float v = (acc[r] + cb[n]) * bnsc[n] + bnbi[n];
    int i = n * 64 + x;                                     // flat [2R, M] index
    int mm = i >> 5, rr = (i & 31) >> 1, cc = i & 1;        // reshape (M, R, 2)
    outA[(((long)(b * 64 + mm) * 16) + rr) * 2 + cc] = v;
  }
}

// ----------------------------- Newton-Schulz polar factor of 64x16 complex matrices

__global__ __launch_bounds__(256) void k_polar(const float* Ain, float* Xout) {
  int b = blockIdx.x, tid = threadIdx.x;
  __shared__ float X[64][16][2];
  __shared__ float Xn[64][16][2];
  __shared__ float G[16][16][2];
  __shared__ float red[256];
  float local = 0.f;
  for (int e = tid; e < 2048; e += 256) {
    float v = Ain[(long)b * 2048 + e];
    ((float*)X)[e] = v;
    local += v * v;
  }
  red[tid] = local;
  __syncthreads();
  for (int s = 128; s > 0; s >>= 1) {
    if (tid < s) red[tid] += red[tid + s];
    __syncthreads();
  }
  float inv = rsqrtf(red[0] + 1e-30f);                      // ||X||_2 <= ||X||_F = 1
  __syncthreads();
  for (int e = tid; e < 2048; e += 256) ((float*)X)[e] *= inv;
  __syncthreads();
  for (int it = 0; it < 16; ++it) {
    int i = tid >> 4, j = tid & 15;                         // G = X^H X
    float gr = 0.f, gi = 0.f;
    for (int k = 0; k < 64; ++k) {
      float xr = X[k][i][0], xi = X[k][i][1];
      float yr = X[k][j][0], yi = X[k][j][1];
      gr += xr * yr + xi * yi;
      gi += xr * yi - xi * yr;
    }
    G[i][j][0] = (i == j ? 1.5f : 0.f) - 0.5f * gr;         // H = 1.5 I - 0.5 G
    G[i][j][1] = -0.5f * gi;
    __syncthreads();
#pragma unroll
    for (int q = 0; q < 4; ++q) {                           // X <- X * H
      int e = tid + q * 256;
      int k = e >> 4, jj = e & 15;
      float ar = 0.f, ai = 0.f;
      for (int ii = 0; ii < 16; ++ii) {
        float xr = X[k][ii][0], xi = X[k][ii][1];
        float hr = G[ii][jj][0], hi = G[ii][jj][1];
        ar += xr * hr - xi * hi;
        ai += xr * hi + xi * hr;
      }
      Xn[k][jj][0] = ar; Xn[k][jj][1] = ai;
    }
    __syncthreads();
    for (int e = tid; e < 2048; e += 256) ((float*)X)[e] = ((float*)Xn)[e];
    __syncthreads();
  }
  for (int e = tid; e < 2048; e += 256) Xout[(long)b * 2048 + e] = ((float*)X)[e];
}

// ----------------------------- sort singular values desc, permute U/V, emit outputs

__global__ __launch_bounds__(256) void k_emit(const float* S, const float* Uo, const float* Vo,
                                              float* out) {
  int b = blockIdx.x, tid = threadIdx.x;
  __shared__ float sv[16];
  __shared__ int ord[16];
  if (tid < 16) sv[tid] = S[b * 16 + tid];
  __syncthreads();
  if (tid < 16) {
    float me = sv[tid];
    int rank = 0;
    for (int j = 0; j < 16; ++j)
      if (sv[j] > me || (sv[j] == me && j < tid)) ++rank;   // stable descending
    ord[rank] = tid;
  }
  __syncthreads();
  float* Uout = out;
  float* Sout = out + 64 * 64 * 16 * 2;
  float* Vout = Sout + 64 * 16;
  if (tid < 16) Sout[b * 16 + tid] = sv[ord[tid]];
  for (int e = tid; e < 64 * 16; e += 256) {
    int m = e >> 4, p = e & 15, j = ord[p];
    long src = (((long)(b * 64 + m)) * 16 + j) * 2;
    long dst = (((long)(b * 64 + m)) * 16 + p) * 2;
    Uout[dst] = Uo[src]; Uout[dst + 1] = Uo[src + 1];
    Vout[dst] = Vo[src]; Vout[dst + 1] = Vo[src + 1];
  }
}

// ================================================================ launcher

extern "C" void kernel_launch(void* const* d_in, const int* in_sizes, int n_in,
                              void* d_out, int out_size, void* d_ws, size_t ws_size,
                              hipStream_t stream) {
  (void)in_sizes; (void)n_in; (void)out_size; (void)ws_size;
  const float* H       = (const float*)d_in[0];
  const float* w1_rb1  = (const float*)d_in[1];
  const float* w2_rb1  = (const float*)d_in[6];
  const float* wsc_rb1 = (const float*)d_in[11];
  const float* w1_rb2  = (const float*)d_in[16];
  const float* w2_rb2  = (const float*)d_in[21];
  const float* wsc_rb2 = (const float*)d_in[26];
  const float* fc1     = (const float*)d_in[31];
  const float* fc2     = (const float*)d_in[32];
  const float* wsa     = (const float*)d_in[33];
  const float* wu      = (const float*)d_in[34];
  const float* ub      = (const float*)d_in[35];
  const float* wv      = (const float*)d_in[40];
  const float* vb      = (const float*)d_in[41];
  const float* wss     = (const float*)d_in[46];
  const float* sbb     = (const float*)d_in[47];

  char* ws = (char*)d_ws;
  size_t off = 0;
  auto alloc = [&](size_t bytes) -> void* {
    void* p = ws + off;
    off += (bytes + 255) & ~(size_t)255;
    return p;
  };

  // packed weights (f16, [N][K])
  h16* w2rb1T = (h16*)alloc(32 * 288 * 2);
  h16* w1rb2T = (h16*)alloc(64 * 288 * 2);
  h16* w2rb2T = (h16*)alloc(64 * 576 * 2);
  h16* wscT   = (h16*)alloc(64 * 32 * 2);
  h16* wuT    = (h16*)alloc(32 * 64 * 2);
  h16* wvT    = (h16*)alloc(32 * 64 * 2);
  // BN scale/bias (f32)
  float* bn1aS = (float*)alloc(32 * 4); float* bn1aB = (float*)alloc(32 * 4);
  float* bn1bS = (float*)alloc(32 * 4); float* bn1bB = (float*)alloc(32 * 4);
  float* bnscS = (float*)alloc(32 * 4); float* bnscB = (float*)alloc(32 * 4);
  float* bn2aS = (float*)alloc(64 * 4); float* bn2aB = (float*)alloc(64 * 4);
  float* bn2bS = (float*)alloc(64 * 4); float* bn2bB = (float*)alloc(64 * 4);
  float* bnsc2S = (float*)alloc(64 * 4); float* bnsc2B = (float*)alloc(64 * 4);
  float* bnuS = (float*)alloc(32 * 4); float* bnuB = (float*)alloc(32 * 4);
  float* bnvS = (float*)alloc(32 * 4); float* bnvB = (float*)alloc(32 * 4);
  float* bnsS = (float*)alloc(16 * 4); float* bnsB = (float*)alloc(16 * 4);
  // activations
  const long P32 = 64L * 64 * 64 * 32, P64 = 64L * 64 * 64 * 64;
  h16* act1  = (h16*)alloc(P32 * 2);
  h16* sc1   = (h16*)alloc(P32 * 2);
  h16* act2  = (h16*)alloc(P32 * 2);
  h16* act3  = (h16*)alloc(P64 * 2);
  h16* sc2   = (h16*)alloc(P64 * 2);
  h16* feat  = (h16*)alloc(P64 * 2);
  h16* featA = (h16*)alloc(P64 * 2);
  float* ca   = (float*)alloc(64 * 64 * 4);
  float* smap = (float*)alloc(64L * 4096 * 2 * 4);
  h16* rowm = (h16*)alloc(64L * 64 * 64 * 2);
  h16* colm = (h16*)alloc(64L * 64 * 64 * 2);
  float* gm  = (float*)alloc(64 * 64 * 4);
  float* U_A = (float*)alloc(64L * 64 * 16 * 2 * 4);
  float* V_A = (float*)alloc(64L * 64 * 16 * 2 * 4);
  float* U_O = (float*)alloc(64L * 64 * 16 * 2 * 4);
  float* V_O = (float*)alloc(64L * 64 * 16 * 2 * 4);
  float* Sb  = (float*)alloc(64 * 16 * 4);

  // ---- prep
  k_bnprep<<<1, 64, 0, stream>>>((const float*)d_in[2],  (const float*)d_in[3],
                                 (const float*)d_in[4],  (const float*)d_in[5],  bn1aS, bn1aB, 32);
  k_bnprep<<<1, 64, 0, stream>>>((const float*)d_in[7],  (const float*)d_in[8],
                                 (const float*)d_in[9],  (const float*)d_in[10], bn1bS, bn1bB, 32);
  k_bnprep<<<1, 64, 0, stream>>>((const float*)d_in[12], (const float*)d_in[13],
                                 (const float*)d_in[14], (const float*)d_in[15], bnscS, bnscB, 32);
  k_bnprep<<<1, 64, 0, stream>>>((const float*)d_in[17], (const float*)d_in[18],
                                 (const float*)d_in[19], (const float*)d_in[20], bn2aS, bn2aB, 64);
  k_bnprep<<<1, 64, 0, stream>>>((const float*)d_in[22], (const float*)d_in[23],
                                 (const float*)d_in[24], (const float*)d_in[25], bn2bS, bn2bB, 64);
  k_bnprep<<<1, 64, 0, stream>>>((const float*)d_in[27], (const float*)d_in[28],
                                 (const float*)d_in[29], (const float*)d_in[30], bnsc2S, bnsc2B, 64);
  k_bnprep<<<1, 64, 0, stream>>>((const float*)d_in[36], (const float*)d_in[37],
                                 (const float*)d_in[38], (const float*)d_in[39], bnuS, bnuB, 32);
  k_bnprep<<<1, 64, 0, stream>>>((const float*)d_in[42], (const float*)d_in[43],
                                 (const float*)d_in[44], (const float*)d_in[45], bnvS, bnvB, 32);
  k_bnprep<<<1, 64, 0, stream>>>((const float*)d_in[48], (const float*)d_in[49],
                                 (const float*)d_in[50], (const float*)d_in[51], bnsS, bnsB, 16);
  k_pack3x3<<<36, 256, 0, stream>>>(w2_rb1, w2rb1T, 32, 32);
  k_pack3x3<<<72, 256, 0, stream>>>(w1_rb2, w1rb2T, 64, 32);
  k_pack3x3<<<144, 256, 0, stream>>>(w2_rb2, w2rb2T, 64, 64);
  k_cvt<<<8, 256, 0, stream>>>(wsc_rb2, wscT, 64 * 32);
  k_cvt<<<8, 256, 0, stream>>>(wu, wuT, 32 * 64);
  k_cvt<<<8, 256, 0, stream>>>(wv, wvT, 32 * 64);

  // ---- backbone
  k_rb1a<<<32768, 256, 0, stream>>>(H, w1_rb1, wsc_rb1, bn1aS, bn1aB, bnscS, bnscB, act1, sc1);
  k_rb1b<<<4096, 256, 0, stream>>>(act1, sc1, w2rb1T, bn1bS, bn1bB, act2);
  k_rb2a<<<4096, 256, 0, stream>>>(act2, w1rb2T, wscT, bn2aS, bn2aB, bnsc2S, bnsc2B, act3, sc2);
  k_rb2b<<<4096, 256, 0, stream>>>(act3, sc2, w2rb2T, bn2bS, bn2bB, feat);

  // ---- CBAM
  k_ca<<<64, 256, 0, stream>>>(feat, fc1, fc2, ca);
  k_smap<<<1024, 256, 0, stream>>>(feat, ca, smap);
  k_saap<<<1024, 256, 0, stream>>>(feat, ca, smap, wsa, featA);

  // ---- reductions + heads
  k_rowmean<<<4096, 64, 0, stream>>>(featA, rowm);
  k_colmean<<<4096, 64, 0, stream>>>(featA, colm);
  k_gmean<<<64, 64, 0, stream>>>(featA, gm);
  k_sproj<<<64, 16, 0, stream>>>(gm, wss, sbb, bnsS, bnsB, Sb);
  k_uv<<<64, 256, 0, stream>>>(rowm, wuT, ub, bnuS, bnuB, U_A);
  k_uv<<<64, 256, 0, stream>>>(colm, wvT, vb, bnvS, bnvB, V_A);

  // ---- polar factors + sort/emit
  k_polar<<<64, 256, 0, stream>>>(U_A, U_O);
  k_polar<<<64, 256, 0, stream>>>(V_A, V_O);
  k_emit<<<64, 256, 0, stream>>>(Sb, U_O, V_O, (float*)d_out);
}
